// MultiHeadedAttention_52140902974312
// MI455X (gfx1250) — compile-verified, matrix operand order checked by reading, not run on hardware
//
#include <hip/hip_runtime.h>

// ---------------------------------------------------------------------------
// MHA forward for MI455X (gfx1250): end-to-end bf16 dataflow, WMMA GEMMs,
// flash attention, CDNA5 async global->LDS staging.
// ---------------------------------------------------------------------------

#define B_ 4
#define S_ 2048
#define H_ 16
#define DM_ 1024
#define DH_ 64

typedef __attribute__((ext_vector_type(16))) __bf16 v16bf;
typedef __attribute__((ext_vector_type(8)))  float  v8f;
typedef int async_v4i __attribute__((vector_size(16)));   // builtin's int4 type

union Frag16 { v16bf v; uint4 u[2]; };
union U128   { uint4 u; unsigned short s[8]; };

__device__ __forceinline__ unsigned short f2bf(float f) {
  return __builtin_bit_cast(unsigned short, static_cast<__bf16>(f));
}

// ---- CDNA5 async global->LDS (ASYNCcnt) with safe fallback ----------------
#if defined(__gfx1250__) && __has_builtin(__builtin_amdgcn_global_load_async_to_lds_b128)
#define CDNA5_ASYNC 1
#else
#define CDNA5_ASYNC 0
#endif

__device__ __forceinline__ void cp16(const unsigned short* g, unsigned short* l) {
#if CDNA5_ASYNC
  __builtin_amdgcn_global_load_async_to_lds_b128(
      (__attribute__((address_space(1))) async_v4i*)g,
      (__attribute__((address_space(3))) async_v4i*)l, 0, 0);
#else
  *reinterpret_cast<uint4*>(l) = *reinterpret_cast<const uint4*>(g);
#endif
}

__device__ __forceinline__ void async_wait_all() {
#if CDNA5_ASYNC
#if __has_builtin(__builtin_amdgcn_s_wait_asynccnt)
  __builtin_amdgcn_s_wait_asynccnt(0);
#else
  asm volatile("s_wait_asynccnt 0x0" ::: "memory");
#endif
#endif
}

// Load a 16x32 bf16 WMMA fragment row for this lane from an LDS row pointer.
// VGPR0-3 hold K = khalf*8 .. +7, VGPR4-7 hold K = 16+khalf*8 .. +7.
__device__ __forceinline__ v16bf load_frag(const unsigned short* row, int koff) {
  Frag16 f;
  f.u[0] = *reinterpret_cast<const uint4*>(row + koff);
  f.u[1] = *reinterpret_cast<const uint4*>(row + koff + 16);
  return f.v;
}

// ---------------------------------------------------------------------------
// Kernel 0: fp32 -> bf16 elementwise (activations + weights), float4 granular.
// ---------------------------------------------------------------------------
__global__ __launch_bounds__(256) void cvt_f32_bf16(
    const float* __restrict__ in, unsigned short* __restrict__ out, int n4)
{
  int i = blockIdx.x * 256 + threadIdx.x;
  if (i >= n4) return;
  float4 v = reinterpret_cast<const float4*>(in)[i];
  union { unsigned short s[4]; uint2 u; } o;
  o.s[0] = f2bf(v.x); o.s[1] = f2bf(v.y); o.s[2] = f2bf(v.z); o.s[3] = f2bf(v.w);
  reinterpret_cast<uint2*>(out)[i] = o.u;
}

// ---------------------------------------------------------------------------
// Kernel 1: C[M,N] = A[M,K] @ B[K,N], bf16 in, bf16 out (Cb) or fp32+resid (Cf).
// Block tile 128x128, K-step 32, double-buffered LDS, 8 waves (wave32).
// ---------------------------------------------------------------------------
#define BM 128
#define BN 128
#define BK 32
#define LDA 40   // 32 + 8 pad, rows stay 16B-aligned

__global__ __launch_bounds__(256) void wmma_gemm_bf16(
    const unsigned short* __restrict__ A, const unsigned short* __restrict__ Bm,
    const float* __restrict__ resid, float* __restrict__ Cf,
    unsigned short* __restrict__ Cb, int M, int N, int K)
{
  __shared__ unsigned short As[2][BM * LDA];
  __shared__ unsigned short Bs[2][BN * LDA];   // transposed: [n][k]

  const int tid  = threadIdx.x;
  const int lane = tid & 31;
  const int wave = tid >> 5;
  const int bm = blockIdx.y * BM;
  const int bn = blockIdx.x * BN;
  const int wm = (wave & 3) * 32;
  const int wn = (wave >> 2) * 64;
  const int lm = lane & 15;
  const int khalf = lane >> 4;

  v8f acc[2][4];
#pragma unroll
  for (int i = 0; i < 2; ++i)
#pragma unroll
    for (int j = 0; j < 4; ++j) acc[i][j] = (v8f)(0.0f);

  auto load_tiles = [&](int kt, int buf) {
    const int k0 = kt * BK;
    // A tile: 128x32 bf16 = 512 x 16B, async copies
#pragma unroll
    for (int p = 0; p < 2; ++p) {
      int idx = p * 256 + tid;
      int r = idx >> 2;
      int c = (idx & 3) * 8;
      cp16(&A[(size_t)(bm + r) * K + k0 + c], &As[buf][r * LDA + c]);
    }
    // B tile: 32x128 bf16, transpose: pack 2 k-rows per b32 LDS store
    {
      int kp = (tid >> 4) * 2;        // k pair 0..30
      int nc = (tid & 15) * 8;        // n group 0..120
      U128 u0, u1;
      u0.u = *reinterpret_cast<const uint4*>(&Bm[(size_t)(k0 + kp) * N + bn + nc]);
      u1.u = *reinterpret_cast<const uint4*>(&Bm[(size_t)(k0 + kp + 1) * N + bn + nc]);
#pragma unroll
      for (int i = 0; i < 8; ++i) {
        unsigned val = (unsigned)u0.s[i] | ((unsigned)u1.s[i] << 16);
        *reinterpret_cast<unsigned*>(&Bs[buf][(nc + i) * LDA + kp]) = val;
      }
    }
  };

  const int NK = K / BK;
  load_tiles(0, 0);
  int buf = 0;
  for (int kt = 0; kt < NK; ++kt) {
    async_wait_all();
    __syncthreads();
    if (kt + 1 < NK) load_tiles(kt + 1, buf ^ 1);
    v16bf af[2], bfr[4];
#pragma unroll
    for (int i = 0; i < 2; ++i)
      af[i] = load_frag(&As[buf][(wm + i * 16 + lm) * LDA], khalf * 8);
#pragma unroll
    for (int j = 0; j < 4; ++j)
      bfr[j] = load_frag(&Bs[buf][(wn + j * 16 + lm) * LDA], khalf * 8);
#pragma unroll
    for (int i = 0; i < 2; ++i)
#pragma unroll
      for (int j = 0; j < 4; ++j)
        acc[i][j] = __builtin_amdgcn_wmma_f32_16x16x32_bf16(
            false, af[i], false, bfr[j], (short)0, acc[i][j], false, false);
    buf ^= 1;
  }

  // C tile layout: lane holds (m = r + khalf*8, n = lm) of each 16x16 tile.
#pragma unroll
  for (int i = 0; i < 2; ++i)
#pragma unroll
    for (int j = 0; j < 4; ++j)
#pragma unroll
      for (int r = 0; r < 8; ++r) {
        int row = bm + wm + i * 16 + r + khalf * 8;
        int col = bn + wn + j * 16 + lm;
        float v = acc[i][j][r];
        if (Cb) {
          Cb[(size_t)row * N + col] = f2bf(v);
        } else {
          if (resid) v += resid[(size_t)row * N + col];
          Cf[(size_t)row * N + col] = v;
        }
      }
}

// ---------------------------------------------------------------------------
// Kernel 2: flash attention, bf16 Q/K/V in workspace layout [b*S+s][h*64+d].
// One WG per (b, h, 128-q block); 8 waves, each owns 16 q-rows.
// ---------------------------------------------------------------------------
#define SQ 128
#define SKB 128
#define QLD 72    // 64 + 8
#define VLD 136   // 128 + 8
#define PLD 136

__global__ __launch_bounds__(256) void flash_attn(
    const unsigned short* __restrict__ Qg, const unsigned short* __restrict__ Kg,
    const unsigned short* __restrict__ Vg, const int* __restrict__ maskg,
    unsigned short* __restrict__ Xg)
{
  __shared__ unsigned short Qs[SQ * QLD];
  __shared__ unsigned short Ks[SKB * QLD];
  __shared__ unsigned short Vt[DH_ * VLD];        // V^T: [d][key]
  __shared__ unsigned short Ps[8 * 16 * PLD];     // per-wave P staging

  const int tid  = threadIdx.x;
  const int lane = tid & 31;
  const int wave = tid >> 5;
  const int qb = blockIdx.x, h = blockIdx.y, b = blockIdx.z;
  const int lm = lane & 15;
  const int khalf = lane >> 4;
  const size_t rowbase = (size_t)b * S_ * DM_ + (size_t)h * DH_;

  // Q block -> LDS (async)
#pragma unroll
  for (int g = 0; g < 4; ++g) {
    int idx = g * 256 + tid;
    int r = idx >> 3;
    int c = (idx & 7) * 8;
    cp16(&Qg[rowbase + (size_t)(qb * SQ + r) * DM_ + c], &Qs[r * QLD + c]);
  }

  float mrun[8], lrun[8];
#pragma unroll
  for (int r = 0; r < 8; ++r) { mrun[r] = -3.0e38f; lrun[r] = 0.0f; }
  v8f o[4];
#pragma unroll
  for (int j = 0; j < 4; ++j) o[j] = (v8f)(0.0f);

  const int wq = wave * 16;

  for (int kb = 0; kb < S_ / SKB; ++kb) {
    __syncthreads();
    // K block -> LDS (async), layout [key][d]
#pragma unroll
    for (int g = 0; g < 4; ++g) {
      int idx = g * 256 + tid;
      int r = idx >> 3;
      int c = (idx & 7) * 8;
      cp16(&Kg[rowbase + (size_t)(kb * SKB + r) * DM_ + c], &Ks[r * QLD + c]);
    }
    // V block -> LDS transposed [d][key], 2 key-rows packed per b32 store
#pragma unroll
    for (int g = 0; g < 2; ++g) {
      int idx = g * 256 + tid;
      int r2 = (idx >> 3) * 2;       // key pair 0..126
      int dg = (idx & 7) * 8;        // d group 0..56
      U128 u0, u1;
      u0.u = *reinterpret_cast<const uint4*>(
          &Vg[rowbase + (size_t)(kb * SKB + r2) * DM_ + dg]);
      u1.u = *reinterpret_cast<const uint4*>(
          &Vg[rowbase + (size_t)(kb * SKB + r2 + 1) * DM_ + dg]);
#pragma unroll
      for (int i = 0; i < 8; ++i) {
        unsigned val = (unsigned)u0.s[i] | ((unsigned)u1.s[i] << 16);
        *reinterpret_cast<unsigned*>(&Vt[(dg + i) * VLD + r2]) = val;
      }
    }
    async_wait_all();
    __syncthreads();

    // scores: 16 q-rows x 128 keys, K-dim = 64 (2 WMMA steps)
    v8f s[8];
#pragma unroll
    for (int j = 0; j < 8; ++j) s[j] = (v8f)(0.0f);
#pragma unroll
    for (int kk = 0; kk < 2; ++kk) {
      v16bf aq = load_frag(&Qs[(wq + lm) * QLD], kk * 32 + khalf * 8);
#pragma unroll
      for (int j = 0; j < 8; ++j) {
        v16bf bk = load_frag(&Ks[(j * 16 + lm) * QLD], kk * 32 + khalf * 8);
        s[j] = __builtin_amdgcn_wmma_f32_16x16x32_bf16(
            false, aq, false, bk, (short)0, s[j], false, false);
      }
    }

    // scale + mask + per-row block max (row lives in a 16-lane half)
    const int qg0 = qb * SQ + wq + khalf * 8;
    float bmax[8];
#pragma unroll
    for (int r = 0; r < 8; ++r) bmax[r] = -3.0e38f;
#pragma unroll
    for (int j = 0; j < 8; ++j) {
      int kgl = kb * SKB + j * 16 + lm;
#pragma unroll
      for (int r = 0; r < 8; ++r) {
        int mv = maskg[((size_t)b * S_ + (qg0 + r)) * (size_t)S_ + kgl];
        float sv = s[j][r] * 0.125f;           // 1/sqrt(64)
        sv = (mv == 0) ? -1.0e9f : sv;
        s[j][r] = sv;
        bmax[r] = fmaxf(bmax[r], sv);
      }
    }
#pragma unroll
    for (int r = 0; r < 8; ++r) {
      float v = bmax[r];
#pragma unroll
      for (int off = 1; off < 16; off <<= 1)
        v = fmaxf(v, __shfl_xor(v, off, 32));
      bmax[r] = v;
    }

    // online softmax update
    float alpha[8], rsum[8];
#pragma unroll
    for (int r = 0; r < 8; ++r) {
      float mn = fmaxf(mrun[r], bmax[r]);
      alpha[r] = __expf(mrun[r] - mn);
      mrun[r] = mn;
      rsum[r] = 0.0f;
    }
#pragma unroll
    for (int j = 0; j < 8; ++j)
#pragma unroll
      for (int r = 0; r < 8; ++r) {
        float p = __expf(s[j][r] - mrun[r]);
        s[j][r] = p;
        rsum[r] += p;
      }
#pragma unroll
    for (int r = 0; r < 8; ++r) {
      float v = rsum[r];
#pragma unroll
      for (int off = 1; off < 16; off <<= 1)
        v += __shfl_xor(v, off, 32);
      lrun[r] = lrun[r] * alpha[r] + v;
    }
#pragma unroll
    for (int j = 0; j < 4; ++j)
#pragma unroll
      for (int r = 0; r < 8; ++r) o[j][r] *= alpha[r];

    // stage P (C layout) to LDS row-major so it re-reads as A fragments
    unsigned short* pw = &Ps[wave * 16 * PLD];
#pragma unroll
    for (int j = 0; j < 8; ++j) {
      int col = j * 16 + lm;
#pragma unroll
      for (int r = 0; r < 8; ++r)
        pw[(r + khalf * 8) * PLD + col] = f2bf(s[j][r]);
    }
    __syncthreads();

    // o += P[16x128] @ V[128x64]  (4 WMMA K-steps x 4 d-tiles)
#pragma unroll
    for (int kk = 0; kk < 4; ++kk) {
      v16bf ap = load_frag(&pw[lm * PLD], kk * 32 + khalf * 8);
#pragma unroll
      for (int j = 0; j < 4; ++j) {
        v16bf bv = load_frag(&Vt[(j * 16 + lm) * VLD], kk * 32 + khalf * 8);
        o[j] = __builtin_amdgcn_wmma_f32_16x16x32_bf16(
            false, ap, false, bv, (short)0, o[j], false, false);
      }
    }
  }

  // normalize and write X[b*S+q][h*64+d] as bf16
#pragma unroll
  for (int j = 0; j < 4; ++j) {
    int d = j * 16 + lm;
#pragma unroll
    for (int r = 0; r < 8; ++r) {
      int qg = qb * SQ + wq + khalf * 8 + r;
      Xg[rowbase + (size_t)qg * DM_ + d] = f2bf(o[j][r] / (lrun[r] + 1e-30f));
    }
  }
}

// ---------------------------------------------------------------------------
// Kernel 3: in-place LayerNorm over rows of Y[8192][1024] (fp32).
// ---------------------------------------------------------------------------
__global__ __launch_bounds__(256) void layernorm_kernel(
    float* __restrict__ Y, const float* __restrict__ gam,
    const float* __restrict__ bet)
{
  __shared__ float red[2][8];
  const int row = blockIdx.x, tid = threadIdx.x;
  const int lane = tid & 31, wave = tid >> 5;
  float4 v = *reinterpret_cast<const float4*>(&Y[(size_t)row * DM_ + tid * 4]);
  float sum = v.x + v.y + v.z + v.w;
  float ssq = v.x * v.x + v.y * v.y + v.z * v.z + v.w * v.w;
#pragma unroll
  for (int off = 1; off < 32; off <<= 1) {
    sum += __shfl_xor(sum, off, 32);
    ssq += __shfl_xor(ssq, off, 32);
  }
  if (lane == 0) { red[0][wave] = sum; red[1][wave] = ssq; }
  __syncthreads();
  sum = red[0][lane & 7];
  ssq = red[1][lane & 7];
#pragma unroll
  for (int off = 1; off < 8; off <<= 1) {
    sum += __shfl_xor(sum, off, 32);
    ssq += __shfl_xor(ssq, off, 32);
  }
  float mu  = sum * (1.0f / DM_);
  float var = ssq * (1.0f / DM_) - mu * mu;
  float inv = rsqrtf(var + 1e-6f);
  float4 g4 = *reinterpret_cast<const float4*>(&gam[tid * 4]);
  float4 b4 = *reinterpret_cast<const float4*>(&bet[tid * 4]);
  float4 o4;
  o4.x = (v.x - mu) * inv * g4.x + b4.x;
  o4.y = (v.y - mu) * inv * g4.y + b4.y;
  o4.z = (v.z - mu) * inv * g4.z + b4.z;
  o4.w = (v.w - mu) * inv * g4.w + b4.w;
  *reinterpret_cast<float4*>(&Y[(size_t)row * DM_ + tid * 4]) = o4;
}

// ---------------------------------------------------------------------------
extern "C" void kernel_launch(void* const* d_in, const int* in_sizes, int n_in,
                              void* d_out, int out_size, void* d_ws, size_t ws_size,
                              hipStream_t stream) {
  (void)in_sizes; (void)n_in; (void)out_size; (void)ws_size;
  const float* query = (const float*)d_in[0];
  const float* key   = (const float*)d_in[1];
  const float* value = (const float*)d_in[2];
  const int*   mask  = (const int*)d_in[3];
  const float* w_qs  = (const float*)d_in[4];
  const float* w_ks  = (const float*)d_in[5];
  const float* w_vs  = (const float*)d_in[6];
  const float* w_out = (const float*)d_in[7];
  const float* ln_g  = (const float*)d_in[8];
  const float* ln_b  = (const float*)d_in[9];
  float* out = (float*)d_out;

  const size_t act  = (size_t)B_ * S_ * DM_;   // 8M elems
  const size_t wmat = (size_t)DM_ * DM_;       // 1M elems
  unsigned short* qb = (unsigned short*)d_ws;  // bf16 workspace
  unsigned short* kb = qb + act;
  unsigned short* vb = kb + act;
  unsigned short* wq = vb + act;
  unsigned short* wk = wq + wmat;
  unsigned short* wv = wk + wmat;
  unsigned short* wo = wv + wmat;
  unsigned short* Qp = wo + wmat;
  unsigned short* Kp = Qp + act;
  unsigned short* Vp = Kp + act;
  unsigned short* Xp = qb;   // qb is dead after the first projection GEMM

  const int M = B_ * S_;
  const int ga = (int)(act / 4 / 256), gw = (int)(wmat / 4 / 256);
  cvt_f32_bf16<<<ga, 256, 0, stream>>>(query, qb, (int)(act / 4));
  cvt_f32_bf16<<<ga, 256, 0, stream>>>(key,   kb, (int)(act / 4));
  cvt_f32_bf16<<<ga, 256, 0, stream>>>(value, vb, (int)(act / 4));
  cvt_f32_bf16<<<gw, 256, 0, stream>>>(w_qs,  wq, (int)(wmat / 4));
  cvt_f32_bf16<<<gw, 256, 0, stream>>>(w_ks,  wk, (int)(wmat / 4));
  cvt_f32_bf16<<<gw, 256, 0, stream>>>(w_vs,  wv, (int)(wmat / 4));
  cvt_f32_bf16<<<gw, 256, 0, stream>>>(w_out, wo, (int)(wmat / 4));

  dim3 gg(DM_ / BN, M / BM);
  wmma_gemm_bf16<<<gg, 256, 0, stream>>>(qb, wq, nullptr, nullptr, Qp, M, DM_, DM_);
  wmma_gemm_bf16<<<gg, 256, 0, stream>>>(kb, wk, nullptr, nullptr, Kp, M, DM_, DM_);
  wmma_gemm_bf16<<<gg, 256, 0, stream>>>(vb, wv, nullptr, nullptr, Vp, M, DM_, DM_);

  flash_attn<<<dim3(S_ / SQ, H_, B_), 256, 0, stream>>>(Qp, Kp, Vp, mask, Xp);

  wmma_gemm_bf16<<<gg, 256, 0, stream>>>(Xp, wo, query, out, nullptr, M, DM_, DM_);
  layernorm_kernel<<<M, 256, 0, stream>>>(out, ln_g, ln_b);
}
